// DocSenModel_49752901157460
// MI455X (gfx1250) — matrix operand order repaired and checked by reading, not compiled, not run on hardware
//
#include <hip/hip_runtime.h>
#include <hip/hip_bf16.h>

// ---------------------------------------------------------------------------
// DocSenModel forward for MI455X (gfx1250, wave32, WMMA bf16 + TDM).
// ---------------------------------------------------------------------------

typedef __attribute__((ext_vector_type(16))) __bf16          v16bf;
typedef __attribute__((ext_vector_type(8)))  float           v8f;
typedef __attribute__((ext_vector_type(16))) unsigned short  u16x16;
typedef __attribute__((ext_vector_type(4)))  float           f32x4;
typedef __attribute__((ext_vector_type(4)))  unsigned int    u32x4;
typedef __attribute__((ext_vector_type(8)))  int             i32x8;
typedef __attribute__((ext_vector_type(4)))  int             i32x4;

constexpr int HDIM = 50;     // hidden size
constexpr int EDIM = 300;    // embedding dim (K)
constexpr int NSEN = 200;    // sentences
constexpr int NWRD = 40;     // words / sentence
constexpr int SW   = NSEN * NWRD;   // 8000 rows (M)
constexpr int KP   = 320;    // K padded to multiple of 32
constexpr int NP   = 64;     // N padded to multiple of 16
constexpr int COUT = 5;

// ---------------------------------------------------------------------------
// Pack W_word [H=50, E=300] f32  ->  Bp [NP=64][KP=320] bf16 (N-major, padded)
// ---------------------------------------------------------------------------
__global__ void k_pack_b(const float* __restrict__ W_word,
                         unsigned short* __restrict__ Bp) {
    int t = blockIdx.x * blockDim.x + threadIdx.x;
    if (t >= NP * KP) return;
    int n = t / KP, k = t % KP;
    float v = (n < HDIM && k < EDIM) ? W_word[n * EDIM + k] : 0.0f;
    Bp[t] = __builtin_bit_cast(unsigned short, (__bf16)v);   // native cvt, RNE
}

// ---------------------------------------------------------------------------
// Fused embedding-gather + word-linear GEMM:
//   x[row, n] = sum_k emb[doc[row], k] * W_word[n, k] + b_word[n]
// One wave per 16-row M tile; 4 accumulators cover N = 0..63 (stores masked
// to n < 50). The 40 KB bf16 B panel is staged into LDS once per block via
// the Tensor Data Mover (tensor_load_to_lds, TENSORcnt fence), so the fully
// unrolled K loop reads B with ds_load_b128 and keeps vmem free for A.
// ---------------------------------------------------------------------------
__global__ void __launch_bounds__(128)
k_word_gemm(const int* __restrict__ doc,
            const float* __restrict__ emb,
            const unsigned short* __restrict__ Bp,
            const float* __restrict__ b_word,
            float* __restrict__ x) {
    __shared__ unsigned short Bs[NP * KP];                   // 40,960 B

    const int lane  = threadIdx.x & 31;
    const int wave  = threadIdx.x >> 5;
    const int tileM = blockIdx.x * 4 + wave;                 // 0..499
    const int rowBase = tileM * 16;
    const int m   = lane & 15;
    const int hi  = lane >> 4;                               // 0: lanes 0-15, 1: 16-31
    const int n0  = lane & 15;

    const int idx = doc[rowBase + m];
    const float* __restrict__ er = emb + (long)idx * EDIM;
    __builtin_prefetch(er, 0, 1);                            // global_prefetch_b8

    // ---- stage B panel into LDS: one TDM descriptor, wave 0 issues ----
#if __has_builtin(__builtin_amdgcn_tensor_load_to_lds)
    if (threadIdx.x < 32) {
        const unsigned long long ga = (unsigned long long)(uintptr_t)Bp;
        const unsigned ldsAddr = (unsigned)(uintptr_t)&Bs[0];
        // D# group 0: count=1 | lds_addr | global_addr[56:0] | type=2
        u32x4 g0 = { 1u,
                     ldsAddr,
                     (unsigned)(ga & 0xFFFFFFFFu),
                     (unsigned)((ga >> 32) & 0x01FFFFFFu) | (2u << 30) };
        // D# group 1: wg_mask=0, data_size=1(2B); tensor 320x64; tile 320x64;
        //             dim0_stride=320; dim1_stride unused (2-D tile).
        i32x8 g1 = { (int)(1u << 16),                        // data_size = 2 bytes
                     (int)((unsigned)KP << 16),              // tensor_dim0[15:0] @63:48
                     (int)((unsigned)NP << 16),              // tensor_dim1[15:0] @95:80
                     (int)((unsigned)KP << 16),              // tile_dim0 @127:112
                     (int)NP,                                // tile_dim1 @143:128
                     (int)KP,                                // tensor_dim0_stride lo32
                     0, 0 };
        i32x4 z4 = { 0, 0, 0, 0 };
#if __clang_major__ >= 23
        i32x8 z8 = { 0, 0, 0, 0, 0, 0, 0, 0 };
        __builtin_amdgcn_tensor_load_to_lds(g0, g1, z4, z4, z8, 0);
#else
        __builtin_amdgcn_tensor_load_to_lds(g0, g1, z4, z4, 0);
#endif
        __builtin_amdgcn_s_wait_tensorcnt(0);                // s_wait_tensorcnt 0
    }
#else
    for (int i = threadIdx.x; i < NP * KP / 8; i += 128)     // fallback: coop copy
        ((uint4*)Bs)[i] = ((const uint4*)Bp)[i];
#endif
    __syncthreads();

    v8f acc[4] = {};

#pragma unroll
    for (int kk = 0; kk < KP; kk += 32) {
        // ---- A fragment (16x32 bf16), contiguous K = kk+16*hi+s ----
        v16bf a;
        const int kA0 = kk + 16 * hi;                        // K of elems 0..7
        const int kA1 = kA0 + 8;                             // K of elems 8..15
        if (kk < 288) {                                      // in-bounds: b128 loads
            const f32x4 f0 = *(const f32x4*)(er + kA0);
            const f32x4 f1 = *(const f32x4*)(er + kA0 + 4);
            const f32x4 f2 = *(const f32x4*)(er + kA1);
            const f32x4 f3 = *(const f32x4*)(er + kA1 + 4);
#pragma unroll
            for (int e = 0; e < 4; ++e) {
                a[e]      = (__bf16)f0[e];                   // native f32->bf16 cvt
                a[4 + e]  = (__bf16)f1[e];
                a[8 + e]  = (__bf16)f2[e];
                a[12 + e] = (__bf16)f3[e];
            }
        } else {                                             // tail K 288..319
#pragma unroll
            for (int e = 0; e < 8; ++e) {
                int ka = kA0 + e, kb = kA1 + e;
                a[e]     = (ka < EDIM) ? (__bf16)er[ka] : (__bf16)0.f;
                a[8 + e] = (kb < EDIM) ? (__bf16)er[kb] : (__bf16)0.f;
            }
        }

        // ---- B fragments from LDS (32x16 bf16): lane = col n0, K = kk+16*hi ----
        const int bofs = kk + 16 * hi;
#pragma unroll
        for (int nt = 0; nt < 4; ++nt) {
            const v16bf b = __builtin_bit_cast(
                v16bf, *(const u16x16*)(Bs + (n0 + 16 * nt) * KP + bofs));
            acc[nt] = __builtin_amdgcn_wmma_f32_16x16x32_bf16(
                false, a, false, b, (short)0, acc[nt], false, false);
        }
    }

    // ---- epilogue: bias + masked store (N < 50) ----
#pragma unroll
    for (int nt = 0; nt < 4; ++nt) {
        const int n = n0 + 16 * nt;
        if (n < HDIM) {
            const float bw = b_word[n];
#pragma unroll
            for (int r = 0; r < 8; ++r) {
                const int mm = r + 8 * hi;                   // D layout: M = r + 8*hi
                x[(rowBase + mm) * HDIM + n] = acc[nt][r] + bw;
            }
        }
    }
}

// ---------------------------------------------------------------------------
// Sentence rep: mean over kernel sizes 1,2,3 of tanh(conv -> avgpool).
// All pooled means derive from one per-(s,h) pass: total and 4 edge words.
// ---------------------------------------------------------------------------
__global__ void k_conv_rep(const float* __restrict__ x,
                           const float* __restrict__ Wc1, const float* __restrict__ bc1,
                           const float* __restrict__ Wc2, const float* __restrict__ bc2,
                           const float* __restrict__ Wc3, const float* __restrict__ bc3,
                           float* __restrict__ rep) {
    __shared__ float ms[6][HDIM];
    const int s = blockIdx.x;
    const int t = threadIdx.x;

    if (t < HDIM) {
        const float* xs = x + (s * NWRD) * HDIM + t;
        float sum = 0.f, x0 = 0.f, x1 = 0.f, x38 = 0.f, x39 = 0.f;
#pragma unroll
        for (int w = 0; w < NWRD; ++w) {
            float v = xs[w * HDIM];
            sum += v;
            if (w == 0)  x0  = v;
            if (w == 1)  x1  = v;
            if (w == 38) x38 = v;
            if (w == 39) x39 = v;
        }
        ms[0][t] = sum * (1.f / 40.f);                     // k=1, j=0
        ms[1][t] = (sum - x39) * (1.f / 39.f);             // k=2, j=0
        ms[2][t] = (sum - x0)  * (1.f / 39.f);             // k=2, j=1
        ms[3][t] = (sum - x38 - x39) * (1.f / 38.f);       // k=3, j=0
        ms[4][t] = (sum - x0  - x39) * (1.f / 38.f);       // k=3, j=1
        ms[5][t] = (sum - x0  - x1)  * (1.f / 38.f);       // k=3, j=2
    }
    __syncthreads();

    if (t < HDIM) {
        float c1 = bc1[t], c2 = bc2[t], c3 = bc3[t];
        for (int h = 0; h < HDIM; ++h) {
            c1 += Wc1[t * HDIM + h] * ms[0][h];
            c2 += Wc2[(t * HDIM + h) * 2 + 0] * ms[1][h]
                + Wc2[(t * HDIM + h) * 2 + 1] * ms[2][h];
            c3 += Wc3[(t * HDIM + h) * 3 + 0] * ms[3][h]
                + Wc3[(t * HDIM + h) * 3 + 1] * ms[4][h]
                + Wc3[(t * HDIM + h) * 3 + 2] * ms[5][h];
        }
        rep[s * HDIM + t] = (tanhf(c1) + tanhf(c2) + tanhf(c3)) * (1.f / 3.f);
    }
}

// ---------------------------------------------------------------------------
// Sequential recurrent scan (200 dependent steps) + mean + softmax head.
// Single workgroup; W_i/W_f/W_g staged in LDS (60 KB); lane d owns row d.
// ---------------------------------------------------------------------------
__global__ void __launch_bounds__(128)
k_scan(const float* __restrict__ rep,
       const float* __restrict__ W_i, const float* __restrict__ b_i,
       const float* __restrict__ W_f, const float* __restrict__ b_f,
       const float* __restrict__ W_g, const float* __restrict__ b_g,
       const float* __restrict__ W_out, const float* __restrict__ b_out,
       float* __restrict__ out) {
    __shared__ float Wl[3][HDIM * 2 * HDIM];   // 3 * 5000 f32 = 60 KB
    __shared__ float comp[2 * HDIM];
    __shared__ float hcur[HDIM];
    __shared__ float hsum[HDIM];
    __shared__ float logits[COUT];

    const int t = threadIdx.x;
    for (int i = t; i < 3 * HDIM * 2 * HDIM; i += blockDim.x) {
        int mtx = i / (HDIM * 2 * HDIM), r = i % (HDIM * 2 * HDIM);
        Wl[mtx][r] = (mtx == 0 ? W_i : mtx == 1 ? W_f : W_g)[r];
    }
    if (t < HDIM) { hcur[t] = 0.f; hsum[t] = 0.f; }
    __syncthreads();

    for (int s = 0; s < NSEN; ++s) {
        if (t < HDIM) {
            comp[t]        = rep[s * HDIM + t];
            comp[HDIM + t] = hcur[t];
        }
        __syncthreads();

        float hn = 0.f;
        if (t < HDIM) {
            float zi = b_i[t], zf = b_f[t], zg = b_g[t];
            const float* wi = &Wl[0][t * 2 * HDIM];
            const float* wf = &Wl[1][t * 2 * HDIM];
            const float* wg = &Wl[2][t * 2 * HDIM];
#pragma unroll 4
            for (int e = 0; e < 2 * HDIM; ++e) {
                const float ce = comp[e];
                zi = fmaf(wi[e], ce, zi);
                zf = fmaf(wf[e], ce, zf);
                zg = fmaf(wg[e], ce, zg);
            }
            const float ig = 1.f / (1.f + __expf(-zi));
            const float fg = 1.f / (1.f + __expf(-zf));
            const float gg = tanhf(zg);
            hn = tanhf(ig * gg + fg * hcur[t]);
        }
        __syncthreads();
        if (t < HDIM) { hcur[t] = hn; hsum[t] += hn; }
        __syncthreads();
    }

    if (t < COUT) {
        float z = b_out[t];
        for (int d = 0; d < HDIM; ++d)
            z = fmaf(W_out[t * HDIM + d], hsum[d] * (1.f / (float)NSEN), z);
        logits[t] = z;
    }
    __syncthreads();
    if (t == 0) {
        float mx = logits[0];
        for (int c = 1; c < COUT; ++c) mx = fmaxf(mx, logits[c]);
        float ex[COUT], den = 0.f;
        for (int c = 0; c < COUT; ++c) { ex[c] = __expf(logits[c] - mx); den += ex[c]; }
        for (int c = 0; c < COUT; ++c) out[c] = ex[c] / den;
    }
}

// ---------------------------------------------------------------------------
extern "C" void kernel_launch(void* const* d_in, const int* in_sizes, int n_in,
                              void* d_out, int out_size, void* d_ws, size_t ws_size,
                              hipStream_t stream) {
    const int*   doc    = (const int*)  d_in[0];
    const float* emb    = (const float*)d_in[1];
    const float* W_word = (const float*)d_in[2];
    const float* b_word = (const float*)d_in[3];
    const float* Wc1    = (const float*)d_in[4];
    const float* bc1    = (const float*)d_in[5];
    const float* Wc2    = (const float*)d_in[6];
    const float* bc2    = (const float*)d_in[7];
    const float* Wc3    = (const float*)d_in[8];
    const float* bc3    = (const float*)d_in[9];
    const float* W_i    = (const float*)d_in[10];
    const float* b_i    = (const float*)d_in[11];
    const float* W_f    = (const float*)d_in[12];
    const float* b_f    = (const float*)d_in[13];
    const float* W_g    = (const float*)d_in[14];
    const float* b_g    = (const float*)d_in[15];
    const float* W_out  = (const float*)d_in[16];
    const float* b_out  = (const float*)d_in[17];
    float* out = (float*)d_out;

    char* ws = (char*)d_ws;
    unsigned short* Bp  = (unsigned short*)ws;                   // 64*320*2   = 40,960 B
    float*          x   = (float*)(ws + 40960);                  // 8000*50*4  = 1,600,000 B
    float*          rep = (float*)(ws + 40960 + 1600000);        // 200*50*4   = 40,000 B

    k_pack_b   <<<(NP * KP + 255) / 256, 256, 0, stream>>>(W_word, Bp);
    k_word_gemm<<<SW / 64, 128, 0, stream>>>(doc, emb, Bp, b_word, x);  // 125 blocks x 4 waves
    k_conv_rep <<<NSEN, 64, 0, stream>>>(x, Wc1, bc1, Wc2, bc2, Wc3, bc3, rep);
    k_scan     <<<1, 128, 0, stream>>>(rep, W_i, b_i, W_f, b_f, W_g, b_g, W_out, b_out, out);
}
